// MultiHeadAttentionChar_50663434223689
// MI455X (gfx1250) — matrix operand order repaired and checked by reading, not compiled, not run on hardware
//
#include <hip/hip_runtime.h>
#include <hip/hip_bf16.h>

typedef __attribute__((ext_vector_type(16))) _Float16 v16h;
typedef __attribute__((ext_vector_type(8)))  _Float16 v8h;
typedef __attribute__((ext_vector_type(8)))  float    v8f;

#define B_  4
#define T_  2048
#define D_  1024
#define H_  16
#define HD_ 64

#define WMMA_F32_F16(a,b,c) \
  __builtin_amdgcn_wmma_f32_16x16x32_f16(false,(a),false,(b),(short)0,(c),false,false)

// ---- WMMA fragment loaders (CDNA5 wave32 16-bit layouts, ISA §7.12.2) ----
// A 16x32 (MxK): lane L holds row M=L%16; elements 0..7 -> K = 8*hi + 0..7,
// elements 8..15 -> K = 8*hi + 16..23  (hi = L/16).
__device__ __forceinline__ v16h load_A16x32(const _Float16* __restrict__ base,
                                            int ld, int lane) {
  const int m  = lane & 15;
  const int kb = (lane >> 4) << 3;          // 0 or 8
  const _Float16* p = base + m * ld + kb;
  v8h lo = *(const v8h*)(p);
  v8h hi = *(const v8h*)(p + 16);
  v16h a;
#pragma unroll
  for (int i = 0; i < 8; ++i) { a[i] = lo[i]; a[i + 8] = hi[i]; }
  return a;
}

// B 32x16 (KxN): lane L holds col N=L%16; elements 0..7 -> K = 8*hi + 0..7,
// elements 8..15 -> K = 8*hi + 16..23. Requires source where consecutive K are contiguous.
__device__ __forceinline__ v16h load_B32x16(const _Float16* __restrict__ base,
                                            int ld, int lane) {
  const int n  = lane & 15;
  const int kb = (lane >> 4) << 3;          // 0 or 8
  const _Float16* p = base + n * ld + kb;
  v8h lo = *(const v8h*)(p);
  v8h hi = *(const v8h*)(p + 16);
  v16h b;
#pragma unroll
  for (int i = 0; i < 8; ++i) { b[i] = lo[i]; b[i + 8] = hi[i]; }
  return b;
}

// ---- CDNA5 async global->LDS (ASYNCcnt-tracked DMA into LDS, ISA §08) ----
// Each lane copies 16B from its global address to its LDS byte offset.
__device__ __forceinline__ void async_copy_b128(unsigned lds_byte_off,
                                                const void* gaddr) {
  asm volatile("global_load_async_to_lds_b128 %0, %1, off"
               :: "v"(lds_byte_off), "v"(gaddr) : "memory");
}
__device__ __forceinline__ void wait_asynccnt_le2() {
  asm volatile("s_wait_asynccnt 0x2" ::: "memory");
}
__device__ __forceinline__ unsigned lds_off_u32(const void* p) {
  // generic LDS pointer low 32 bits = LDS byte address (aperture mapping)
  return (unsigned)(size_t)p;
}

// ---------------- prep: casts / transposes ----------------
__global__ void cast_f32_to_f16_kernel(const float* __restrict__ src,
                                       _Float16* __restrict__ dst, int n) {
  int i = blockIdx.x * blockDim.x + threadIdx.x;
  if (i < n) dst[i] = (_Float16)src[i];
}

// src [H][D][HD] -> dst [H][HD][D]   (so B fragments load k-contiguous)
__global__ void transpose_cast_w_kernel(const float* __restrict__ src,
                                        _Float16* __restrict__ dst) {
  int i = blockIdx.x * blockDim.x + threadIdx.x;   // over H*D*HD
  int e = i % HD_;
  int d = (i / HD_) % D_;
  int h = i / (D_ * HD_);
  dst[(h * HD_ + e) * D_ + d] = (_Float16)src[i];
}

// ---------------- QKV projection (WMMA) ----------------
__global__ __launch_bounds__(256) void qkv_kernel(
    const _Float16* __restrict__ x16,
    const _Float16* __restrict__ wqt,
    const _Float16* __restrict__ wkt,
    const _Float16* __restrict__ wvt,
    const float*    __restrict__ s_raw,
    _Float16* __restrict__ q16,
    _Float16* __restrict__ k16,
    _Float16* __restrict__ vt16) {
  const int lane  = threadIdx.x & 31;
  const int wid   = blockIdx.x * (blockDim.x >> 5) + (threadIdx.x >> 5);
  const int tTile = wid & 127;
  const int h     = (wid >> 7) & 15;
  const int b     = (wid >> 11) & 3;
  const int mat   = wid >> 13;                    // 0=q, 1=k, 2=v

  const _Float16* wt = (mat == 0) ? wqt : (mat == 1) ? wkt : wvt;
  const _Float16* xa = x16 + ((size_t)(b * T_ + tTile * 16)) * D_;
  const _Float16* wb = wt + h * (HD_ * D_);

  v8f acc[4] = {};
  for (int kk = 0; kk < D_; kk += 32) {
    v16h a = load_A16x32(xa + kk, D_, lane);
    __builtin_prefetch(xa + kk + 32, 0, 3);
#pragma unroll
    for (int nt = 0; nt < 4; ++nt) {
      v16h bm = load_B32x16(wb + (nt * 16) * D_ + kk, D_, lane);
      acc[nt] = WMMA_F32_F16(a, bm, acc[nt]);
    }
  }

  const int n  = lane & 15;
  const int hi = lane >> 4;
  const int bh = b * H_ + h;
  if (mat == 2) {
    // store V transposed: vt16[(bh*HD + e)*T + t]
#pragma unroll
    for (int nt = 0; nt < 4; ++nt)
#pragma unroll
      for (int r = 0; r < 8; ++r) {
        int t = tTile * 16 + r + 8 * hi;
        int e = nt * 16 + n;
        vt16[((size_t)(bh * HD_ + e)) * T_ + t] = (_Float16)acc[nt][r];
      }
  } else {
    _Float16* dst = (mat == 0) ? q16 : k16;
    float sp = 0.f;
    if (mat == 0) {
      float s = s_raw[h];
      sp = (s > 20.f) ? s : log1pf(__expf(s));     // softplus
    }
#pragma unroll
    for (int nt = 0; nt < 4; ++nt)
#pragma unroll
      for (int r = 0; r < 8; ++r) {
        int t = tTile * 16 + r + 8 * hi;
        int e = nt * 16 + n;
        float v = acc[nt][r];
        // fold ssmax scale and HD^-0.5 into q
        if (mat == 0) v *= sp * __logf((float)(t + 1)) * 0.125f;
        dst[((size_t)(bh * T_ + t)) * HD_ + e] = (_Float16)v;
      }
  }
}

// ---------------- causal flash attention (WMMA + async LDS staging) --------
// Block = 8 waves sharing (b,h), consecutive t-tiles. K/V chunks staged once
// per block into LDS via async DMA, double-buffered against WMMA compute.
__global__ __launch_bounds__(256) void flash_attn_kernel(
    const _Float16* __restrict__ q16,
    const _Float16* __restrict__ k16,
    const _Float16* __restrict__ vt16,
    _Float16* __restrict__ ao16) {
  __shared__ __align__(16) _Float16 kbuf[2][32 * HD_];   // [s_local][e]
  __shared__ __align__(16) _Float16 vbuf[2][HD_ * 32];   // [e][s_local]
  __shared__ __align__(16) _Float16 pbuf[8][16 * 32];    // per-wave P staging

  const int tid     = threadIdx.x;
  const int lane    = tid & 31;
  const int wslot   = tid >> 5;
  const int ttGroup = blockIdx.x & 15;
  const int h       = (blockIdx.x >> 4) & 15;
  const int b       = blockIdx.x >> 8;
  const int bh      = b * H_ + h;
  const int tTile   = ttGroup * 8 + wslot;
  const int tBase   = tTile * 16;
  const int n       = lane & 15;
  const int hi      = lane >> 4;

  const int nChunk = 4 * (ttGroup + 1);   // block-uniform chunk count

  // per-thread staging slices: K chunk = 4KB contiguous; V chunk = 64 rows x 64B
  const unsigned kdst[2] = { lds_off_u32((char*)kbuf[0] + tid * 16),
                             lds_off_u32((char*)kbuf[1] + tid * 16) };
  const int ve    = tid >> 2;            // 0..63
  const int vpart = tid & 3;             // 0..3 (16B each)
  const unsigned vdst[2] = { lds_off_u32((char*)vbuf[0] + ve * 64 + vpart * 16),
                             lds_off_u32((char*)vbuf[1] + ve * 64 + vpart * 16) };
  const _Float16* ksrc0 = k16 + (size_t)bh * T_ * HD_ + tid * 8;      // +sBase*HD
  const _Float16* vsrc0 = vt16 + ((size_t)(bh * HD_ + ve)) * T_ + vpart * 8;  // +sBase

  // Q fragments: HD=64 -> two 16x32 A tiles, loaded once
  const _Float16* qbase = q16 + ((size_t)(bh * T_ + tBase)) * HD_;
  v16h aq0 = load_A16x32(qbase + 0,  HD_, lane);
  v16h aq1 = load_A16x32(qbase + 32, HD_, lane);

  v8f o[4] = {};
  float mrow[8], lrow[8];
#pragma unroll
  for (int r = 0; r < 8; ++r) { mrow[r] = -1e30f; lrow[r] = 0.f; }

  _Float16* pb = pbuf[wslot];

  // prologue: stage chunk 0 into buffer 0
  async_copy_b128(kdst[0], ksrc0);
  async_copy_b128(vdst[0], vsrc0);

  for (int ch = 0; ch < nChunk; ++ch) {
    const int sBase = ch * 32;
    // issue next chunk into the other buffer (clamped -> constant asynccnt)
    const int nx = (ch + 1 < nChunk) ? ch + 1 : ch;
    async_copy_b128(kdst[(ch + 1) & 1], ksrc0 + (size_t)(nx * 32) * HD_);
    async_copy_b128(vdst[(ch + 1) & 1], vsrc0 + nx * 32);
    wait_asynccnt_le2();          // chunk ch resident (this wave's share)
    __syncthreads();              // all waves' shares resident

    const _Float16* kc = kbuf[ch & 1];
    const _Float16* vc = vbuf[ch & 1];

    // scalar-gate the causally-needed region so EXEC stays all-1s for WMMA
    int act = (sBase <= tBase + 15) ? 1 : 0;
    if (__builtin_amdgcn_readfirstlane(act)) {
      // ---- scores: two 16x16 tiles over this 32-col chunk ----
      v8f c0 = {}, c1 = {};
      {
        v16h b00 = load_B32x16(kc + 0,  HD_, lane);            // cols 0..15
        v16h b01 = load_B32x16(kc + 32, HD_, lane);
        c0 = WMMA_F32_F16(aq0, b00, c0);
        c0 = WMMA_F32_F16(aq1, b01, c0);
        v16h b10 = load_B32x16(kc + 16 * HD_ + 0,  HD_, lane); // cols 16..31
        v16h b11 = load_B32x16(kc + 16 * HD_ + 32, HD_, lane);
        c1 = WMMA_F32_F16(aq0, b10, c1);
        c1 = WMMA_F32_F16(aq1, b11, c1);
      }
      // ---- causal mask + online softmax ----
#pragma unroll
      for (int r = 0; r < 8; ++r) {
        int t = tBase + r + 8 * hi;
        if (sBase + n      > t) c0[r] = -1e30f;
        if (sBase + 16 + n > t) c1[r] = -1e30f;
      }
#pragma unroll
      for (int r = 0; r < 8; ++r) {
        float mc = fmaxf(c0[r], c1[r]);
#pragma unroll
        for (int m = 1; m < 16; m <<= 1) mc = fmaxf(mc, __shfl_xor(mc, m, 32));
        float nm    = fmaxf(mrow[r], mc);
        float alpha = __expf(mrow[r] - nm);
        float p0    = __expf(c0[r] - nm);
        float p1    = __expf(c1[r] - nm);
        float ps    = p0 + p1;
#pragma unroll
        for (int m = 1; m < 16; m <<= 1) ps += __shfl_xor(ps, m, 32);
        lrow[r] = lrow[r] * alpha + ps;
        mrow[r] = nm;
#pragma unroll
        for (int nt = 0; nt < 4; ++nt) o[nt][r] *= alpha;
        // stash P (C-layout row = r+8*hi, cols n / n+16) for transpose
        int row = r + 8 * hi;
        pb[row * 32 + n]      = (_Float16)p0;
        pb[row * 32 + n + 16] = (_Float16)p1;
      }
      // ---- O += P * V (P re-read in A-layout; per-wave LDS is in-order) ----
      v16h ap = load_A16x32(pb, 32, lane);
#pragma unroll
      for (int nt = 0; nt < 4; ++nt) {
        v16h bv = load_B32x16(vc + (nt * 16) * 32, 32, lane);
        o[nt] = WMMA_F32_F16(ap, bv, o[nt]);
      }
    }
    __syncthreads();              // before overwriting buf[ch&1] next iter
  }

  // ---- normalize, write concat-head layout [B,T,D] ----
#pragma unroll
  for (int nt = 0; nt < 4; ++nt)
#pragma unroll
    for (int r = 0; r < 8; ++r) {
      int t = tBase + r + 8 * hi;
      int e = nt * 16 + n;
      ao16[((size_t)(b * T_ + t)) * D_ + h * HD_ + e] =
          (_Float16)(o[nt][r] / lrow[r]);
    }
}

// ---------------- output projection (WMMA) + bias ----------------
__global__ __launch_bounds__(256) void proj_kernel(
    const _Float16* __restrict__ ao16,
    const _Float16* __restrict__ pw16,    // proj_W [o][d] row-major
    const float*    __restrict__ bias,
    float* __restrict__ out) {
  const int lane    = threadIdx.x & 31;
  const int wid     = blockIdx.x * 8 + (threadIdx.x >> 5);
  const int rowTile = wid & 511;
  const int oTile   = wid >> 9;
  const int rowBase = rowTile * 16;
  const int oBase   = oTile * 64;

  const _Float16* abase = ao16 + (size_t)rowBase * D_;
  v8f acc[4] = {};
  for (int kk = 0; kk < D_; kk += 32) {
    v16h a = load_A16x32(abase + kk, D_, lane);
    __builtin_prefetch(abase + kk + 32, 0, 3);
#pragma unroll
    for (int nt = 0; nt < 4; ++nt) {
      v16h bm = load_B32x16(pw16 + ((size_t)(oBase + nt * 16)) * D_ + kk, D_, lane);
      acc[nt] = WMMA_F32_F16(a, bm, acc[nt]);
    }
  }
  const int n = lane & 15, hi = lane >> 4;
#pragma unroll
  for (int nt = 0; nt < 4; ++nt) {
    float bv = bias[oBase + nt * 16 + n];
#pragma unroll
    for (int r = 0; r < 8; ++r) {
      int row = rowBase + r + 8 * hi;
      out[(size_t)row * D_ + oBase + nt * 16 + n] = acc[nt][r] + bv;
    }
  }
}

// ---------------- host launcher ----------------
extern "C" void kernel_launch(void* const* d_in, const int* in_sizes, int n_in,
                              void* d_out, int out_size, void* d_ws, size_t ws_size,
                              hipStream_t stream) {
  const float* x      = (const float*)d_in[0];
  const float* Wq     = (const float*)d_in[1];
  const float* Wk     = (const float*)d_in[2];
  const float* Wv     = (const float*)d_in[3];
  const float* s_raw  = (const float*)d_in[4];
  const float* proj_W = (const float*)d_in[5];
  const float* proj_b = (const float*)d_in[6];
  float* out = (float*)d_out;

  // workspace carve-up (all f16)
  _Float16* ws = (_Float16*)d_ws;
  size_t off = 0;
  const size_t NX = (size_t)B_ * T_ * D_;     // 8M
  const size_t NW = (size_t)H_ * D_ * HD_;    // 1M
  _Float16* x16  = ws + off; off += NX;
  _Float16* wqt  = ws + off; off += NW;
  _Float16* wkt  = ws + off; off += NW;
  _Float16* wvt  = ws + off; off += NW;
  _Float16* pw16 = ws + off; off += (size_t)D_ * D_;
  _Float16* q16  = ws + off; off += NX;
  _Float16* k16  = ws + off; off += NX;
  _Float16* vt16 = ws + off; off += NX;
  _Float16* ao16 = ws + off; off += NX;

  // 1) casts / transposes
  cast_f32_to_f16_kernel<<<(int)(NX / 256), 256, 0, stream>>>(x, x16, (int)NX);
  transpose_cast_w_kernel<<<(int)(NW / 256), 256, 0, stream>>>(Wq, wqt);
  transpose_cast_w_kernel<<<(int)(NW / 256), 256, 0, stream>>>(Wk, wkt);
  transpose_cast_w_kernel<<<(int)(NW / 256), 256, 0, stream>>>(Wv, wvt);
  cast_f32_to_f16_kernel<<<(D_ * D_) / 256, 256, 0, stream>>>(proj_W, pw16, D_ * D_);

  // 2) QKV projections: 3*B*H*(T/16) = 24576 waves / 8 per block
  qkv_kernel<<<3072, 256, 0, stream>>>(x16, wqt, wkt, wvt, s_raw, q16, k16, vt16);

  // 3) causal flash attention: one block per (b,h,8 t-tiles) = 1024 blocks
  flash_attn_kernel<<<1024, 256, 0, stream>>>(q16, k16, vt16, ao16);

  // 4) output projection: (B*T/16)*(D/64) = 8192 waves
  proj_kernel<<<1024, 256, 0, stream>>>(ao16, pw16, proj_b, out);
}
